// DeepseekV3MoE_15728170238343
// MI455X (gfx1250) — matrix operand order repaired and checked by reading, not compiled, not run on hardware
//
#include <hip/hip_runtime.h>

// ---------------- problem constants (from reference setup) ----------------
constexpr int Tn   = 2048;   // tokens = B*S
constexpr int Hn   = 1024;   // hidden
constexpr int En   = 16;     // routed experts
constexpr int In   = 512;    // moe intermediate
constexpr int SIn  = 1024;   // shared intermediate
constexpr int TOPK = 4;

typedef __attribute__((ext_vector_type(16))) __bf16 v16bf;
typedef __attribute__((ext_vector_type(8)))  float  v8f;

// ---------------- WMMA helpers (CDNA5 16x16x32 bf16, wave32) ----------------
__device__ inline v8f wmma_bf16(v16bf a, v16bf b, v8f c) {
  // (neg_a, A, neg_b, B, c_mod, C, reuse_a, reuse_b)
  return __builtin_amdgcn_wmma_f32_16x16x32_bf16(false, a, false, b, (short)0, c,
                                                 false, false);
}

// A-matrix 16x32 bf16 from an LDS tile laid out row-major [rows][pitch],
// starting at row mbase. ISA: lanes 0-15 -> M=lane, K pairs; lanes 16-31 same M,
// K offset +8; VGPRs 4..7 hold K+16.
__device__ inline v16bf load_A_lds(const __bf16* a, int mbase, int kbase, int pitch,
                                   int lane) {
  v16bf r;
  const int m  = mbase + (lane & 15);
  const int kb = kbase + ((lane & 16) ? 8 : 0);
#pragma unroll
  for (int p = 0; p < 4; ++p) {
    r[2 * p]         = a[m * pitch + kb + 2 * p];
    r[2 * p + 1]     = a[m * pitch + kb + 2 * p + 1];
    r[8 + 2 * p]     = a[m * pitch + kb + 16 + 2 * p];
    r[8 + 2 * p + 1] = a[m * pitch + kb + 16 + 2 * p + 1];
  }
  return r;
}

// B-matrix 32x16 raw f32 fetch from row-major weight [K][ldn].
// ISA B layout: lane N = lane%16; lanes 0-15 hold K=0..7,16..23, lanes 16-31 K=8..15,24..31;
// element i (after conversion) holds K = kb + i (+8 for i>=8). Raw-first loading lets the
// compiler clause all 16 dword loads and decouple the wait from the WMMA.
__device__ inline void loadB_raw(float* __restrict__ r, const float* __restrict__ w,
                                 int kbase, int nbase, int ldn, int lane) {
  const int n  = nbase + (lane & 15);
  const int kb = kbase + ((lane & 16) ? 8 : 0);
#pragma unroll
  for (int p = 0; p < 16; ++p) r[p] = w[(size_t)(kb + p + ((p >= 8) ? 8 : 0)) * ldn + n];
}

__device__ inline v16bf cvtB(const float* __restrict__ r) {
  v16bf b;
#pragma unroll
  for (int p = 0; p < 16; ++p) b[p] = (__bf16)r[p];  // v_cvt_pk_bf16_f32 pairs
  return b;
}

__device__ inline float silu_f(float v) { return v / (1.f + __expf(-v)); }

// ---------------- kernel 0: clear per-expert counters ----------------
__global__ void moe_zero_cnt(int* cnt) {
  if (threadIdx.x < En) cnt[threadIdx.x] = 0;
}

// ---------------- kernel 1: gating (one wave32 per token) ----------------
__global__ __launch_bounds__(256) void moe_gate(const float* __restrict__ x,
                                                const float* __restrict__ gw,
                                                int* __restrict__ cnt,
                                                int* __restrict__ idx,
                                                float* __restrict__ wgt,
                                                __bf16* __restrict__ xbf) {
  const int lane = threadIdx.x & 31;
  const int t    = (blockIdx.x * blockDim.x + threadIdx.x) >> 5;  // token per wave
  if (t >= Tn) return;

  float xr[32];
  const float* xp = x + (size_t)t * Hn;
#pragma unroll
  for (int j = 0; j < 32; ++j) {
    float v = xp[j * 32 + lane];
    xr[j]   = v;
    xbf[(size_t)t * Hn + j * 32 + lane] = (__bf16)v;
  }

  float s[En];
#pragma unroll
  for (int e = 0; e < En; ++e) {
    const float* gp = gw + (size_t)e * Hn;
    float acc = 0.f;
#pragma unroll
    for (int j = 0; j < 32; ++j) acc = fmaf(xr[j], gp[j * 32 + lane], acc);
#pragma unroll
    for (int off = 16; off > 0; off >>= 1) acc += __shfl_xor(acc, off, 32);
    s[e] = 1.f / (1.f + __expf(-acc));
  }

  int sel[TOPK];
  float sw[TOPK];
  float sum = 0.f;
  unsigned taken = 0;
#pragma unroll
  for (int k = 0; k < TOPK; ++k) {
    float best = -1.f;
    int be = 0;
#pragma unroll
    for (int e = 0; e < En; ++e)
      if (!((taken >> e) & 1u) && s[e] > best) { best = s[e]; be = e; }
    taken |= 1u << be;
    sel[k] = be;
    sw[k]  = best;
    sum += best;
  }
  const float inv = 1.f / (sum + 1e-20f);

  if (lane == 0) {
#pragma unroll
    for (int k = 0; k < TOPK; ++k) {
      int e   = sel[k];
      int pos = atomicAdd(&cnt[e], 1);
      idx[e * Tn + pos] = t;
      wgt[e * Tn + pos] = sw[k] * inv;
    }
  }
}

// ---------------- kernel 2: shared expert (writes d_out, clears poison) ------
// 32-token tile per block, 16 waves in two m-groups of 8. Both m-groups stream
// the same B addresses -> second read is WGP$-resident, halving L2 traffic.
__global__ __launch_bounds__(512) void moe_shared(const __bf16* __restrict__ xbf,
                                                  const float* __restrict__ wg,
                                                  const float* __restrict__ wu,
                                                  const float* __restrict__ wd,
                                                  float* __restrict__ out) {
  __shared__ __bf16 xt[32 * Hn];    // 64 KB
  __shared__ __bf16 act[32 * SIn];  // 64 KB
  const int t0  = blockIdx.x * 32;
  const int tid = threadIdx.x;

  // stage x tile with b128 copies (rows contiguous; pitch == Hn)
  for (int i = tid; i < 32 * (Hn / 8); i += 512) {
    const int m = i >> 7;  // Hn/8 = 128 uint4 per row
    const int c = i & 127;
    ((uint4*)xt)[i] = ((const uint4*)(xbf + (size_t)(t0 + m) * Hn))[c];
  }
  __syncthreads();

  const int wave = tid >> 5, lane = tid & 31;
  const int mg = (wave >> 3) * 16;  // m-group row base: 0 or 16
  const int wl = wave & 7;          // lane-group within m-group

  // ---- gate/up GEMMs: [32xH] @ [HxSI] in two phases of 512 cols ----
  for (int ph = 0; ph < 2; ++ph) {
    const int n0 = ph * 512 + wl * 64;
    v8f cg[4] = {}, cu[4] = {};
    v16bf a = load_A_lds(xt, mg, 0, Hn, lane);
    float rg[16], ru[16];
    loadB_raw(rg, wg, 0, n0, SIn, lane);
    loadB_raw(ru, wu, 0, n0, SIn, lane);
    for (int k = 0; k < Hn; k += 32) {
      const int kn = (k + 32 < Hn) ? k + 32 : k;
      const v16bf acur = a;
#pragma unroll
      for (int q = 0; q < 4; ++q) {
        float ng[16], nu[16];
        const int qn = (q + 1) & 3;
        const int kq = (q == 3) ? kn : k;
        loadB_raw(ng, wg, kq, n0 + 16 * qn, SIn, lane);  // prefetch next tile
        loadB_raw(nu, wu, kq, n0 + 16 * qn, SIn, lane);
        if (q == 3) a = load_A_lds(xt, mg, kn, Hn, lane);
        cg[q] = wmma_bf16(acur, cvtB(rg), cg[q]);
        cu[q] = wmma_bf16(acur, cvtB(ru), cu[q]);
#pragma unroll
        for (int i = 0; i < 16; ++i) { rg[i] = ng[i]; ru[i] = nu[i]; }
      }
    }
#pragma unroll
    for (int q = 0; q < 4; ++q) {
      const int nn = n0 + 16 * q + (lane & 15);
#pragma unroll
      for (int r = 0; r < 8; ++r) {
        const int m = mg + ((lane < 16) ? r : r + 8);
        act[m * SIn + nn] = (__bf16)(silu_f(cg[q][r]) * cu[q][r]);
      }
    }
  }
  __syncthreads();

  // ---- down GEMM: [32xSI] @ [SIxH], each wave: 1 m-tile x 128 cols of H ----
  const int c0 = wl * 128;
  v8f co[8] = {};
  v16bf a2 = load_A_lds(act, mg, 0, SIn, lane);
  float rd[16];
  loadB_raw(rd, wd, 0, c0, Hn, lane);
  for (int k = 0; k < SIn; k += 32) {
    const int kn = (k + 32 < SIn) ? k + 32 : k;
    const v16bf acur = a2;
#pragma unroll
    for (int q = 0; q < 8; ++q) {
      float nd[16];
      const int qn = (q + 1) & 7;
      const int kq = (q == 7) ? kn : k;
      loadB_raw(nd, wd, kq, c0 + 16 * qn, Hn, lane);
      if (q == 7) a2 = load_A_lds(act, mg, kn, SIn, lane);
      co[q] = wmma_bf16(acur, cvtB(rd), co[q]);
#pragma unroll
      for (int i = 0; i < 16; ++i) rd[i] = nd[i];
    }
  }
#pragma unroll
  for (int q = 0; q < 8; ++q) {
    const int nn = c0 + 16 * q + (lane & 15);
#pragma unroll
    for (int r = 0; r < 8; ++r) {
      const int m = mg + ((lane < 16) ? r : r + 8);
      out[(size_t)(t0 + m) * Hn + nn] = co[q][r];  // plain store: init d_out
    }
  }
}

// ---------------- kernel 3: routed experts (sparse gather + atomic combine) ---
__global__ __launch_bounds__(512) void moe_routed(const __bf16* __restrict__ xbf,
                                                  const float* __restrict__ wg,
                                                  const float* __restrict__ wu,
                                                  const float* __restrict__ wd,
                                                  const int* __restrict__ cnt,
                                                  const int* __restrict__ idx,
                                                  const float* __restrict__ wgt,
                                                  float* __restrict__ out) {
  __shared__ __bf16 xt[32 * Hn];   // 64 KB
  __shared__ __bf16 act[32 * In];  // 32 KB
  __shared__ int   toks[32];
  __shared__ float tw[32];

  const int e    = blockIdx.x >> 6;  // / 64 tiles
  const int tile = blockIdx.x & 63;
  const int ne   = cnt[e];
  if (tile * 32 >= ne) return;  // uniform across block

  const int tid = threadIdx.x;
  if (tid < 32) {
    int pos = tile * 32 + tid;
    if (pos < ne) {
      toks[tid] = idx[e * Tn + pos];
      tw[tid]   = wgt[e * Tn + pos];
    } else {
      toks[tid] = 0;
      tw[tid]   = 0.f;  // padded rows contribute 0 via weight
    }
  }
  __syncthreads();

  for (int i = tid; i < 32 * (Hn / 8); i += 512) {
    const int m = i >> 7;
    const int c = i & 127;
    ((uint4*)xt)[i] = ((const uint4*)(xbf + (size_t)toks[m] * Hn))[c];
  }
  __syncthreads();

  const int wave = tid >> 5, lane = tid & 31;
  const int mg = (wave >> 3) * 16;
  const int wl = wave & 7;
  const float* wge = wg + (size_t)e * Hn * In;
  const float* wue = wu + (size_t)e * Hn * In;
  const float* wde = wd + (size_t)e * In * Hn;

  // ---- gate/up: [32xH] @ [HxI], each wave: 1 m-tile x 64 cols of I ----
  const int n0 = wl * 64;
  v8f cg[4] = {}, cu[4] = {};
  {
    v16bf a = load_A_lds(xt, mg, 0, Hn, lane);
    float rg[16], ru[16];
    loadB_raw(rg, wge, 0, n0, In, lane);
    loadB_raw(ru, wue, 0, n0, In, lane);
    for (int k = 0; k < Hn; k += 32) {
      const int kn = (k + 32 < Hn) ? k + 32 : k;
      const v16bf acur = a;
#pragma unroll
      for (int q = 0; q < 4; ++q) {
        float ng[16], nu[16];
        const int qn = (q + 1) & 3;
        const int kq = (q == 3) ? kn : k;
        loadB_raw(ng, wge, kq, n0 + 16 * qn, In, lane);
        loadB_raw(nu, wue, kq, n0 + 16 * qn, In, lane);
        if (q == 3) a = load_A_lds(xt, mg, kn, Hn, lane);
        cg[q] = wmma_bf16(acur, cvtB(rg), cg[q]);
        cu[q] = wmma_bf16(acur, cvtB(ru), cu[q]);
#pragma unroll
        for (int i = 0; i < 16; ++i) { rg[i] = ng[i]; ru[i] = nu[i]; }
      }
    }
  }
#pragma unroll
  for (int q = 0; q < 4; ++q) {
    const int nn = n0 + 16 * q + (lane & 15);
#pragma unroll
    for (int r = 0; r < 8; ++r) {
      const int m = mg + ((lane < 16) ? r : r + 8);
      act[m * In + nn] = (__bf16)(silu_f(cg[q][r]) * cu[q][r]);
    }
  }
  __syncthreads();

  // ---- down: [32xI] @ [IxH], each wave: 1 m-tile x 128 cols of H ----
  const int c0 = wl * 128;
  v8f co[8] = {};
  {
    v16bf a2 = load_A_lds(act, mg, 0, In, lane);
    float rd[16];
    loadB_raw(rd, wde, 0, c0, Hn, lane);
    for (int k = 0; k < In; k += 32) {
      const int kn = (k + 32 < In) ? k + 32 : k;
      const v16bf acur = a2;
#pragma unroll
      for (int q = 0; q < 8; ++q) {
        float nd[16];
        const int qn = (q + 1) & 7;
        const int kq = (q == 7) ? kn : k;
        loadB_raw(nd, wde, kq, c0 + 16 * qn, Hn, lane);
        if (q == 7) a2 = load_A_lds(act, mg, kn, In, lane);
        co[q] = wmma_bf16(acur, cvtB(rd), co[q]);
#pragma unroll
        for (int i = 0; i < 16; ++i) rd[i] = nd[i];
      }
    }
  }
#pragma unroll
  for (int q = 0; q < 8; ++q) {
    const int nn = c0 + 16 * q + (lane & 15);
#pragma unroll
    for (int r = 0; r < 8; ++r) {
      const int m = mg + ((lane < 16) ? r : r + 8);
      atomicAdd(&out[(size_t)toks[m] * Hn + nn], co[q][r] * tw[m]);
    }
  }
}

// ---------------- launch ----------------
extern "C" void kernel_launch(void* const* d_in, const int* in_sizes, int n_in,
                              void* d_out, int out_size, void* d_ws, size_t ws_size,
                              hipStream_t stream) {
  const float* x   = (const float*)d_in[0];  // [T, H]
  const float* gw  = (const float*)d_in[1];  // [E, H]
  const float* weg = (const float*)d_in[2];  // [E, H, I]
  const float* weu = (const float*)d_in[3];  // [E, H, I]
  const float* wed = (const float*)d_in[4];  // [E, I, H]
  const float* wsg = (const float*)d_in[5];  // [H, SI]
  const float* wsu = (const float*)d_in[6];  // [H, SI]
  const float* wsd = (const float*)d_in[7];  // [SI, H]
  float* out = (float*)d_out;

  // workspace: cnt(256B) | idx(E*T*4) | wgt(E*T*4) | xbf(T*H*2) ~ 4.5 MB
  char*   ws  = (char*)d_ws;
  int*    cnt = (int*)ws;
  int*    idx = (int*)(ws + 256);
  float*  wgt = (float*)(ws + 256 + (size_t)En * Tn * 4);
  __bf16* xbf = (__bf16*)(ws + 256 + 2 * (size_t)En * Tn * 4);

  moe_zero_cnt<<<1, 32, 0, stream>>>(cnt);
  moe_gate<<<Tn / 8, 256, 0, stream>>>(x, gw, cnt, idx, wgt, xbf);
  moe_shared<<<Tn / 32, 512, 0, stream>>>(xbf, wsg, wsu, wsd, out);
  moe_routed<<<En * (Tn / 32), 512, 0, stream>>>(xbf, weg, weu, wed, cnt, idx, wgt, out);
}
